// DynGraphBlock_89781996356035
// MI455X (gfx1250) — matrix operand order repaired and checked by reading, not compiled, not run on hardware
//
#include <hip/hip_runtime.h>
#include <hip/hip_bf16.h>

// Problem constants (match reference)
#define NB   256
#define CD   128
#define TD   2048
#define FEPS 1e-6f
#define SQRT_T 45.25483399593904f   // sqrt(2048)
#define LSTR 40                     // LDS row stride in bf16 elems (32 data + 8 pad; 80B rows, 16B aligned)

typedef __attribute__((ext_vector_type(16))) __bf16 v16bf;
typedef __attribute__((ext_vector_type(2)))  __bf16 v2bf;
typedef __attribute__((ext_vector_type(8)))  float  v8f;

union ABOperand { v16bf v; uint4 q[2]; };

// Native f32 -> bf16 (RNE).
__device__ __forceinline__ unsigned short bf16_bits(float f) {
    union { __bf16 h; unsigned short u; } r;
    r.h = (__bf16)f;
    return r.u;
}

// Packed 2xf32 -> 2xbf16 (v_cvt_pk_bf16_f32, confirmed emitted on gfx1250).
__device__ __forceinline__ unsigned int pack_bf16x2(float lo, float hi) {
#if defined(__has_builtin) && __has_builtin(__builtin_amdgcn_cvt_pk_bf16_f32)
    union { v2bf h; unsigned int u; } r;
    r.h = __builtin_amdgcn_cvt_pk_bf16_f32(lo, hi);
    return r.u;
#else
    union { __bf16 h[2]; unsigned int u; } r;
    r.h[0] = (__bf16)lo;
    r.h[1] = (__bf16)hi;
    return r.u;
#endif
}

__device__ __forceinline__ v8f v8f_zero() {
    v8f z;
#pragma unroll
    for (int k = 0; k < 8; ++k) z[k] = 0.0f;
    return z;
}
__device__ __forceinline__ float wave_sum(float v) {
#pragma unroll
    for (int off = 16; off > 0; off >>= 1) v += __shfl_xor(v, off, 32);
    return v;
}
__device__ __forceinline__ float wave_max(float v) {
#pragma unroll
    for (int off = 16; off > 0; off >>= 1) v = fmaxf(v, __shfl_xor(v, off, 32));
    return v;
}

// ---------------------------------------------------------------------------
// Kernel 1: per-row sum / sumsq -> u = s/sqrt(T), q = 1/(sqrt(T)*(std+eps))
// ---------------------------------------------------------------------------
__global__ __launch_bounds__(256) void stats_kernel(const float* __restrict__ x,
                                                    float* __restrict__ u,
                                                    float* __restrict__ q) {
    const int row  = blockIdx.x * 8 + (threadIdx.x >> 5);
    const int lane = threadIdx.x & 31;
    const float* xr = x + (size_t)row * TD;
    float s = 0.0f, ss = 0.0f;
#pragma unroll 4
    for (int t = lane; t < TD; t += 32) {
        float v = xr[t];
        s += v; ss += v * v;
    }
    s  = wave_sum(s);
    ss = wave_sum(ss);
    if (lane == 0) {
        float mu  = s * (1.0f / TD);
        float var = (ss - s * mu) * (1.0f / (TD - 1));
        float sd  = sqrtf(fmaxf(var, 0.0f)) + FEPS;
        u[row] = s * (1.0f / SQRT_T);
        q[row] = 1.0f / (sd * SQRT_T);
    }
}

// ---------------------------------------------------------------------------
// Kernel 2: batched Gram  G[n] = X[n] * X[n]^T  (bf16 WMMA, f32 accum).
// 8 waves per batch. T is walked in 256-wide super-chunks of 8 statically
// unrolled 32-chunks (mirrors zgemm's fully unrolled structure so the
// scheduler keeps two B register sets alive -> partial dscnt waits).
// ---------------------------------------------------------------------------
__global__ __launch_bounds__(256, 1) void gram_kernel(const float* __restrict__ x,
                                                      float* __restrict__ G) {
    __shared__ unsigned short tile[2][CD * LSTR];     // 2 x (128 x 32 bf16, padded)
    const int n    = blockIdx.x;
    const int tid  = threadIdx.x;
    const int wave = tid >> 5;
    const int lane = tid & 31;
    const float* xb = x + (size_t)n * CD * TD;

    v8f acc[8];
#pragma unroll
    for (int i = 0; i < 8; ++i) acc[i] = v8f_zero();

    const int srow = tid >> 1;                  // staging: row, 16-col half
    const int sch  = (tid & 1) * 16;
    const int am   = wave * 16 + (lane & 15);   // A operand row
    const int akb  = (lane >> 4) * 8;           // A operand K base
    const int bn   = lane & 15;                 // B operand col within tile
    const int bkb  = (lane >> 4) * 8;           // B operand K base

    auto stage = [&](int kc, int buf) {
        const float* base = xb + (size_t)srow * TD + kc + sch;
        __builtin_prefetch(base + 64, 0, 1);    // speculative global_prefetch_b8, 2 chunks ahead
        const float4* src = reinterpret_cast<const float4*>(base);
        float4 f0 = src[0], f1 = src[1], f2 = src[2], f3 = src[3];
        uint4* dst = reinterpret_cast<uint4*>(&tile[buf][srow * LSTR + sch]);
        dst[0] = make_uint4(pack_bf16x2(f0.x, f0.y), pack_bf16x2(f0.z, f0.w),
                            pack_bf16x2(f1.x, f1.y), pack_bf16x2(f1.z, f1.w));
        dst[1] = make_uint4(pack_bf16x2(f2.x, f2.y), pack_bf16x2(f2.z, f2.w),
                            pack_bf16x2(f3.x, f3.y), pack_bf16x2(f3.z, f3.w));
    };

    auto compute = [&](int buf) {
        const unsigned short* tl = tile[buf];
        ABOperand a;
        a.q[0] = *reinterpret_cast<const uint4*>(tl + am * LSTR + akb);
        a.q[1] = *reinterpret_cast<const uint4*>(tl + am * LSTR + akb + 16);
        ABOperand b[8];                         // all tiles live -> loads stay in flight
#pragma unroll
        for (int nj = 0; nj < 8; ++nj) {
            const int bc = nj * 16 + bn;
            b[nj].q[0] = *reinterpret_cast<const uint4*>(tl + bc * LSTR + bkb);
            b[nj].q[1] = *reinterpret_cast<const uint4*>(tl + bc * LSTR + bkb + 16);
        }
#pragma unroll
        for (int nj = 0; nj < 8; ++nj)
            acc[nj] = __builtin_amdgcn_wmma_f32_16x16x32_bf16(
                false, a.v, false, b[nj].v, (short)0, acc[nj], false, false);
    };

    stage(0, 0);
    __syncthreads();
    // main: all but the last 256-wide super-chunk, 8 chunks statically unrolled
    for (int sc = 0; sc < TD - 256; sc += 256) {
#pragma unroll
        for (int c = 0; c < 8; ++c) {
            stage(sc + c * 32 + 32, (c + 1) & 1);   // next chunk's global loads first
            compute(c & 1);
            __syncthreads();
        }
    }
    // peeled last super-chunk
#pragma unroll
    for (int c = 0; c < 8; ++c) {
        if (c < 7) stage((TD - 256) + c * 32 + 32, (c + 1) & 1);
        compute(c & 1);
        if (c < 7) __syncthreads();
    }

    // C/D layout: lane L, vgpr v -> (M = v + 8*(L>>4), N = L&15)
    float* Gb = G + (size_t)n * CD * CD;
    const int rbase = wave * 16 + 8 * (lane >> 4);
    const int cbase = lane & 15;
#pragma unroll
    for (int nj = 0; nj < 8; ++nj)
#pragma unroll
        for (int v = 0; v < 8; ++v)
            Gb[(size_t)(rbase + v) * CD + nj * 16 + cbase] = acc[nj][v];
}

// ---------------------------------------------------------------------------
// Kernel 3: corr + relu + top-8 mask, in place on G. One wave per row.
// ---------------------------------------------------------------------------
__global__ __launch_bounds__(256) void topk_kernel(float* __restrict__ G,
                                                   const float* __restrict__ u,
                                                   const float* __restrict__ q) {
    const int row  = blockIdx.x * 8 + (threadIdx.x >> 5);
    const int lane = threadIdx.x & 31;
    const int n = row >> 7, i = row & 127;
    float* Gr = G + ((size_t)n * CD + i) * CD;
    const float* ub = u + n * CD;
    const float* qb = q + n * CD;
    const float ui = ub[i], qi = qb[i];

    float vals[4];
#pragma unroll
    for (int s = 0; s < 4; ++s) {
        const int j = lane + s * 32;
        float a = (Gr[j] - ui * ub[j]) * (qi * qb[j]);
        vals[s] = fmaxf(a, 0.0f);                 // relu
    }
    float cur = 3.4e38f;                          // 8th-largest via wave-max extraction
#pragma unroll
    for (int it = 0; it < 8; ++it) {
        float m = -3.4e38f;
#pragma unroll
        for (int s = 0; s < 4; ++s)
            if (vals[s] < cur) m = fmaxf(m, vals[s]);
        cur = wave_max(m);
    }
#pragma unroll
    for (int s = 0; s < 4; ++s) {
        const int j = lane + s * 32;
        Gr[j] = (vals[s] >= cur) ? vals[s] : 0.0f;
    }
}

// ---------------------------------------------------------------------------
// Kernel 4: symmetrize + self-loop + row-normalize + EMA -> A (f32 + bf16 copy)
// ---------------------------------------------------------------------------
__global__ __launch_bounds__(256) void norm_ema_kernel(const float* __restrict__ M,
                                                       const float* __restrict__ A_prev,
                                                       float* __restrict__ A_out,
                                                       unsigned short* __restrict__ A_bf) {
    const int row  = blockIdx.x * 8 + (threadIdx.x >> 5);
    const int lane = threadIdx.x & 31;
    const int n = row >> 7, i = row & 127;
    const float* Mb = M + (size_t)n * CD * CD;

    float v[4];
    float deg = 0.0f;
#pragma unroll
    for (int s = 0; s < 4; ++s) {
        const int j = lane + s * 32;
        float a = 0.5f * (Mb[(size_t)i * CD + j] + Mb[(size_t)j * CD + i]);
        if (j == i) a += 1.0f;
        v[s] = a;
        deg += a;
    }
    deg = wave_sum(deg);
    const float inv = 1.0f / (deg + FEPS);
    const size_t base = ((size_t)n * CD + i) * CD;
#pragma unroll
    for (int s = 0; s < 4; ++s) {
        const int j = lane + s * 32;
        float af = 0.8f * A_prev[base + j] + 0.2f * (v[s] * inv);
        A_out[base + j] = af;
        A_bf[base + j]  = bf16_bits(af);
    }
}

// ---------------------------------------------------------------------------
// Kernel 5: z = A @ x ; out = x + gamma*z.  bf16 WMMA, f32 accumulate.
// blockIdx.x = 128-col t-chunk, blockIdx.y = batch. 8 waves per block.
// ---------------------------------------------------------------------------
__global__ __launch_bounds__(256, 1) void zgemm_kernel(const float* __restrict__ x,
                                                       const unsigned short* __restrict__ A_bf,
                                                       const float* __restrict__ gamma_p,
                                                       float* __restrict__ out) {
    __shared__ unsigned short xt[2][CD * LSTR];   // 2 x (128 t-rows x 32 j-cols bf16)
    const int n     = blockIdx.y;
    const int tbase = blockIdx.x * 128;
    const int tid   = threadIdx.x;
    const int wave  = tid >> 5;
    const int lane  = tid & 31;
    const float* xb = x + (size_t)n * CD * TD;
    const unsigned short* Ab = A_bf + (size_t)n * CD * CD;

    v8f acc[8];
#pragma unroll
    for (int i = 0; i < 8; ++i) acc[i] = v8f_zero();

    const int sj  = (tid & 15) * 2;             // staging: 2 adjacent channels
    const int st  = (tid >> 4) * 8;             // staging: 8 t columns
    const int ai  = wave * 16 + (lane & 15);
    const int akb = (lane >> 4) * 8;
    const int bn  = lane & 15;
    const int bkb = (lane >> 4) * 8;

    auto stage = [&](int kc, int buf) {
        const float* r0 = xb + (size_t)(kc + sj) * TD + tbase + st;
        const float* r1 = r0 + TD;
        float4 a0 = *reinterpret_cast<const float4*>(r0);
        float4 a1 = *reinterpret_cast<const float4*>(r0 + 4);
        float4 c0 = *reinterpret_cast<const float4*>(r1);
        float4 c1 = *reinterpret_cast<const float4*>(r1 + 4);
        float va[8] = { a0.x, a0.y, a0.z, a0.w, a1.x, a1.y, a1.z, a1.w };
        float vc[8] = { c0.x, c0.y, c0.z, c0.w, c1.x, c1.y, c1.z, c1.w };
#pragma unroll
        for (int m = 0; m < 8; ++m)             // transposed, packed b32 stores
            *reinterpret_cast<unsigned int*>(&xt[buf][(st + m) * LSTR + sj]) =
                pack_bf16x2(va[m], vc[m]);
    };

    auto compute = [&](int buf, int kc) {
        ABOperand a;                            // A straight from global bf16 (L2-resident)
        const unsigned short* ap = Ab + (size_t)ai * CD + kc + akb;
        a.q[0] = *reinterpret_cast<const uint4*>(ap);
        a.q[1] = *reinterpret_cast<const uint4*>(ap + 16);
        const unsigned short* tl = xt[buf];
        ABOperand b[8];                         // all tiles live -> loads stay in flight
#pragma unroll
        for (int tj = 0; tj < 8; ++tj) {
            const int tt = tj * 16 + bn;
            b[tj].q[0] = *reinterpret_cast<const uint4*>(tl + tt * LSTR + bkb);
            b[tj].q[1] = *reinterpret_cast<const uint4*>(tl + tt * LSTR + bkb + 16);
        }
#pragma unroll
        for (int tj = 0; tj < 8; ++tj)
            acc[tj] = __builtin_amdgcn_wmma_f32_16x16x32_bf16(
                false, a.v, false, b[tj].v, (short)0, acc[tj], false, false);
    };

    stage(0, 0);
    __syncthreads();
#pragma unroll
    for (int ck = 0; ck < 4; ++ck) {            // K = 128, 4 chunks of 32
        if (ck < 3) stage((ck + 1) * 32, (ck + 1) & 1);
        compute(ck & 1, ck * 32);
        if (ck < 3) __syncthreads();
    }

    // epilogue: out = x + gamma * z
    const float g = *gamma_p;
    const int rbase = wave * 16 + 8 * (lane >> 4);
    const int cbase = lane & 15;
#pragma unroll
    for (int tj = 0; tj < 8; ++tj) {
#pragma unroll
        for (int v = 0; v < 8; ++v) {
            const size_t idx = ((size_t)n * CD + (rbase + v)) * TD + tbase + tj * 16 + cbase;
            out[idx] = x[idx] + g * acc[tj][v];
        }
    }
}

// ---------------------------------------------------------------------------
extern "C" void kernel_launch(void* const* d_in, const int* in_sizes, int n_in,
                              void* d_out, int out_size, void* d_ws, size_t ws_size,
                              hipStream_t stream) {
    (void)in_sizes; (void)n_in; (void)out_size; (void)ws_size;
    const float* x      = (const float*)d_in[0];
    const float* A_prev = (const float*)d_in[1];
    const float* gamma  = (const float*)d_in[2];

    float* out_x = (float*)d_out;                                // [N,C,T]
    float* out_A = out_x + (size_t)NB * CD * TD;                 // [N,C,C]

    char* ws = (char*)d_ws;
    float* G            = (float*)ws;                            // 16 MB
    float* u            = (float*)(ws + (size_t)16777216);       // 128 KB
    float* q            = (float*)(ws + (size_t)16777216 + 131072);
    unsigned short* Abf = (unsigned short*)(ws + (size_t)17039360); // 8 MB (16B aligned)

    const int rows = NB * CD;                 // 32768 rows, 8 waves/block
    stats_kernel   <<<rows / 8, 256, 0, stream>>>(x, u, q);
    gram_kernel    <<<NB,       256, 0, stream>>>(x, G);
    topk_kernel    <<<rows / 8, 256, 0, stream>>>(G, u, q);
    norm_ema_kernel<<<rows / 8, 256, 0, stream>>>(G, A_prev, out_A, Abf);
    zgemm_kernel   <<<dim3(TD / 128, NB), 256, 0, stream>>>(x, Abf, gamma, out_x);
}
